// MultiHeadAttentionBlock_40561671143470
// MI455X (gfx1250) — matrix operand order repaired and checked by reading, not compiled, not run on hardware
//
#include <hip/hip_runtime.h>

#define BB 2
#define SS 2048
#define DD 1024
#define HH 16
#define DKK 64

typedef __bf16 bf16;
typedef __attribute__((ext_vector_type(16))) __bf16 v16bf;
typedef __attribute__((ext_vector_type(8)))  __bf16 v8bf;
typedef __attribute__((ext_vector_type(8)))  float  v8f;
typedef __attribute__((ext_vector_type(4)))  unsigned int u32x4;
typedef __attribute__((ext_vector_type(8)))  int i32x8;
typedef __attribute__((ext_vector_type(4)))  int i32x4;

union FragA { v16bf v; v8bf h[2]; };

static __device__ __forceinline__ v8f wmma_bf16(v16bf a, v16bf b, v8f c) {
  // D = A(16x32 bf16) * B(32x16 bf16) + C(16x16 f32)
  return __builtin_amdgcn_wmma_f32_16x16x32_bf16(false, a, false, b, (short)0, c,
                                                 false, false);
}

#ifndef __has_builtin
#define __has_builtin(x) 0
#endif
#if __has_builtin(__builtin_amdgcn_tensor_load_to_lds) && \
    __has_builtin(__builtin_amdgcn_s_wait_tensorcnt)
#define HAVE_TDM 1
#else
#define HAVE_TDM 0
#endif

#if HAVE_TDM
// 2D tile load via the Tensor Data Mover.  D# built per CDNA5 ISA §8.3/8.4.
// dims/strides in elements (data_size = 2 bytes).
static __device__ __forceinline__ void
tdm_load_2d(unsigned int lds_addr, unsigned long long gaddr,
            unsigned int tensor_d0, unsigned int tensor_d1,
            unsigned int tile_d0, unsigned int tile_d1, unsigned int stride0) {
  u32x4 g0;
  g0[0] = 1u;                                         // count=1, user descriptor
  g0[1] = lds_addr;                                   // LDS byte address
  g0[2] = (unsigned int)(gaddr & 0xffffffffu);        // global_addr[31:0]
  g0[3] = (unsigned int)((gaddr >> 32) & 0x1ffffffu)  // global_addr[56:32]
          | (2u << 30);                               // type = 2 ("image")
  i32x8 g1;
  g1[0] = (int)(1u << 16);                            // data_size=1 -> 2 bytes
  g1[1] = (int)(tensor_d0 << 16);                     // tensor_dim0[15:0]
  g1[2] = (int)((tensor_d0 >> 16) | (tensor_d1 << 16));
  g1[3] = (int)((tensor_d1 >> 16) | (tile_d0 << 16)); // tile_dim0
  g1[4] = (int)(tile_d1 & 0xffffu);                   // tile_dim1, tile_dim2=0
  g1[5] = (int)stride0;                               // tensor_dim0_stride[31:0]
  g1[6] = 0;
  g1[7] = 0;
  i32x4 z4 = {0, 0, 0, 0};
#if __clang_major__ >= 23
  i32x8 z8 = {0, 0, 0, 0, 0, 0, 0, 0};
  __builtin_amdgcn_tensor_load_to_lds(g0, g1, z4, z4, z8, 0);
#else
  __builtin_amdgcn_tensor_load_to_lds(g0, g1, z4, z4, 0);
#endif
}
#endif

// ---------------------------------------------------------------------------
// fp32 -> bf16 conversion (grid-stride)
// ---------------------------------------------------------------------------
__global__ void cvt_f32_to_bf16(const float* __restrict__ x, bf16* __restrict__ y, int n) {
  int i = blockIdx.x * blockDim.x + threadIdx.x;
  int stride = gridDim.x * blockDim.x;
  for (; i < n; i += stride) y[i] = (bf16)x[i];
}

// ---------------------------------------------------------------------------
// Y = X @ W^T + bias.  X: [M=B*S, K=D] bf16 row-major.  W: [N=D, K=D] bf16
// row-major (nn.Linear weight).  MODE 0: store bf16 head-split [B,H,S,DK].
// MODE 1: store fp32 flat [M, D].
// Block: 256 thr = 8 waves; block tile 256(M) x 64(N); wave tile 32 x 64:
// two A fragments share every B fragment (double arithmetic intensity).
// ---------------------------------------------------------------------------
template <int MODE>
__global__ void __launch_bounds__(256)
proj_gemm(const bf16* __restrict__ X, const bf16* __restrict__ W,
          const float* __restrict__ bias, bf16* __restrict__ outB,
          float* __restrict__ outF) {
  const int lane = threadIdx.x & 31;
  const int wave = threadIdx.x >> 5;
  const int hi   = lane >> 4;        // half-wave select
  const int l15  = lane & 15;
  const int m0   = blockIdx.x * 256 + wave * 32;
  const int n0   = blockIdx.y * 64;
  const int koff  = hi * 8;          // A-fragment K base within 32-chunk
  const int koff2 = hi * 16;         // unused; B fragments also use koff

  v8f acc[2][4] = {};

  const bf16* arow0 = X + (size_t)(m0 + l15) * DD;
  const bf16* arow1 = X + (size_t)(m0 + 16 + l15) * DD;
  for (int k0 = 0; k0 < DD; k0 += 32) {
    FragA a0, a1;
    a0.h[0] = *(const v8bf*)(arow0 + k0 + koff);
    a0.h[1] = *(const v8bf*)(arow0 + k0 + 16 + koff);
    a1.h[0] = *(const v8bf*)(arow1 + k0 + koff);
    a1.h[1] = *(const v8bf*)(arow1 + k0 + 16 + koff);
#pragma unroll
    for (int t = 0; t < 4; ++t) {
      const bf16* brow = W + (size_t)(n0 + t * 16 + l15) * DD + k0 + koff;
      FragA b;
      b.h[0] = *(const v8bf*)(brow);
      b.h[1] = *(const v8bf*)(brow + 16);
      acc[0][t] = wmma_bf16(a0.v, b.v, acc[0][t]);
      acc[1][t] = wmma_bf16(a1.v, b.v, acc[1][t]);
    }
  }

#pragma unroll
  for (int ri = 0; ri < 2; ++ri) {
#pragma unroll
    for (int t = 0; t < 4; ++t) {
      const int n = n0 + t * 16 + l15;
      const float bv = bias[n];
#pragma unroll
      for (int i = 0; i < 8; ++i) {
        const int r = m0 + ri * 16 + i + hi * 8;   // global row (C layout)
        const float val = acc[ri][t][i] + bv;
        if (MODE == 0) {
          const int b_ = r / SS, s = r % SS;
          const int h_ = n >> 6, dk = n & 63;
          outB[(((size_t)(b_ * HH + h_)) * SS + s) * DKK + dk] = (bf16)val;
        } else {
          outF[(size_t)r * DD + n] = val;
        }
      }
    }
  }
}

// ---------------------------------------------------------------------------
// Flash attention: one block = (b,h) x 64 query rows; 4 waves x 16 rows.
// Key blocks of 64 staged into LDS by the Tensor Data Mover (TDM); causal
// mask applied only on the diagonal block.  ctx output: bf16 [B, S, D].
// ---------------------------------------------------------------------------
__global__ void __launch_bounds__(128)
attn_kernel(const bf16* __restrict__ Q, const bf16* __restrict__ K,
            const bf16* __restrict__ V, bf16* __restrict__ ctx) {
  __shared__ __align__(16) bf16 Klds[64 * 64];    // K block, row-major [n][d]
  __shared__ __align__(16) bf16 Vlds[64 * 64];    // V block, row-major [n][d]
  __shared__ __align__(16) bf16 VTlds[64 * 64];   // V^T block [d][n]
  __shared__ __align__(16) bf16 Plds[4][16 * 64]; // per-wave P tile [m][n]

  const int tid  = threadIdx.x;
  const int lane = tid & 31;
  const int wave = tid >> 5;
  const int hi   = lane >> 4;
  const int l15  = lane & 15;
  const int koff  = hi * 8;
  const int koff2 = hi * 16;

  const int q0 = blockIdx.x * 64;
  const int bh = blockIdx.y;
  const int b_ = bh / HH;
  const int h_ = bh % HH;

  const bf16* Qb = Q + (size_t)bh * SS * DKK;
  const bf16* Kb = K + (size_t)bh * SS * DKK;
  const bf16* Vb = V + (size_t)bh * SS * DKK;

  const int qw = q0 + wave * 16;   // first query row of this wave

  // Q A-fragments for both 32-wide k-steps of DK=64
  FragA aq[2];
  {
    const bf16* qrow = Qb + (size_t)(qw + l15) * DKK;
#pragma unroll
    for (int ks = 0; ks < 2; ++ks) {
      aq[ks].h[0] = *(const v8bf*)(qrow + ks * 32 + koff);
      aq[ks].h[1] = *(const v8bf*)(qrow + ks * 32 + 16 + koff);
    }
  }

  float mrun[8], lrun[8];
  v8f o[4] = {};
#pragma unroll
  for (int i = 0; i < 8; ++i) { mrun[i] = -1e30f; lrun[i] = 0.f; }

  for (int kb0 = 0; kb0 <= q0; kb0 += 64) {
    __syncthreads();   // previous iteration's LDS reads done
#if HAVE_TDM
    if (wave == 0) {
      tdm_load_2d((unsigned int)(size_t)&Klds[0],
                  (unsigned long long)(size_t)(Kb + (size_t)kb0 * DKK),
                  DKK, SS, 64, 64, DKK);
      tdm_load_2d((unsigned int)(size_t)&Vlds[0],
                  (unsigned long long)(size_t)(Vb + (size_t)kb0 * DKK),
                  DKK, SS, 64, 64, DKK);
      __builtin_amdgcn_s_wait_tensorcnt(0);
    }
#else
    {
      const int r = tid >> 1;              // 0..63
      const int cseg = (tid & 1) * 32;     // 0 or 32
      const v8bf* ksrc = (const v8bf*)(Kb + (size_t)(kb0 + r) * DKK + cseg);
      v8bf* kdst = (v8bf*)(&Klds[r * 64 + cseg]);
#pragma unroll
      for (int j = 0; j < 4; ++j) kdst[j] = ksrc[j];
      const v8bf* vsrc = (const v8bf*)(Vb + (size_t)(kb0 + r) * DKK + cseg);
      v8bf* vdst = (v8bf*)(&Vlds[r * 64 + cseg]);
#pragma unroll
      for (int j = 0; j < 4; ++j) vdst[j] = vsrc[j];
    }
#endif
    __syncthreads();   // staged tiles visible

    // ---- build V^T in LDS (for P@V B-fragments) ----
    {
      const int n = tid >> 1;
      const int dseg = (tid & 1) * 32;
#pragma unroll
      for (int j = 0; j < 4; ++j) {
        v8bf tmp = *(const v8bf*)(&Vlds[n * 64 + dseg + 8 * j]);
#pragma unroll
        for (int e = 0; e < 8; ++e)
          VTlds[(dseg + 8 * j + e) * 64 + n] = tmp[e];
      }
    }
    __syncthreads();

    // ---- scores S = Q K^T (16x64 per wave, 8 WMMA) ----
    v8f sc[4] = {};
#pragma unroll
    for (int t = 0; t < 4; ++t) {
      const bf16* krow = &Klds[(t * 16 + l15) * 64];
      FragA bk;
      bk.h[0] = *(const v8bf*)(krow + koff);
      bk.h[1] = *(const v8bf*)(krow + koff + 16);
      sc[t] = wmma_bf16(aq[0].v, bk.v, sc[t]);
      bk.h[0] = *(const v8bf*)(krow + 32 + koff);
      bk.h[1] = *(const v8bf*)(krow + 32 + koff + 16);
      sc[t] = wmma_bf16(aq[1].v, bk.v, sc[t]);
    }

    const float scale = 0.125f;            // 1/sqrt(DK)
    const bool need_mask = (kb0 + 63 > qw);
#pragma unroll
    for (int t = 0; t < 4; ++t) {
#pragma unroll
      for (int i = 0; i < 8; ++i) {
        float s = sc[t][i] * scale;
        if (need_mask) {
          const int sq = qw + i + hi * 8;
          const int sk = kb0 + t * 16 + l15;
          if (sk > sq) s = -1e9f;
        }
        sc[t][i] = s;
      }
    }

    // ---- online softmax (row stats across 16-lane groups) ----
    float p[4][8];
#pragma unroll
    for (int i = 0; i < 8; ++i) {
      float bm = fmaxf(fmaxf(sc[0][i], sc[1][i]), fmaxf(sc[2][i], sc[3][i]));
      bm = fmaxf(bm, __shfl_xor(bm, 1, 32));
      bm = fmaxf(bm, __shfl_xor(bm, 2, 32));
      bm = fmaxf(bm, __shfl_xor(bm, 4, 32));
      bm = fmaxf(bm, __shfl_xor(bm, 8, 32));
      const float mnew = fmaxf(mrun[i], bm);
      const float alpha = __expf(mrun[i] - mnew);
      mrun[i] = mnew;
      float rs = 0.f;
#pragma unroll
      for (int t = 0; t < 4; ++t) {
        const float pv = __expf(sc[t][i] - mnew);
        p[t][i] = pv;
        rs += pv;
      }
      rs += __shfl_xor(rs, 1, 32);
      rs += __shfl_xor(rs, 2, 32);
      rs += __shfl_xor(rs, 4, 32);
      rs += __shfl_xor(rs, 8, 32);
      lrun[i] = lrun[i] * alpha + rs;
#pragma unroll
      for (int t = 0; t < 4; ++t) o[t][i] *= alpha;
    }

    // ---- transpose P (C layout -> A layout) through per-wave LDS ----
    bf16* pl = &Plds[wave][0];
#pragma unroll
    for (int t = 0; t < 4; ++t)
#pragma unroll
      for (int i = 0; i < 8; ++i)
        pl[(i + hi * 8) * 64 + t * 16 + l15] = (bf16)p[t][i];
    // LDS ops are in-order within a wave; same-wave RAW is safe.

    FragA ap[2];
    {
      const bf16* prow = pl + l15 * 64;
#pragma unroll
      for (int ks = 0; ks < 2; ++ks) {
        ap[ks].h[0] = *(const v8bf*)(prow + ks * 32 + koff);
        ap[ks].h[1] = *(const v8bf*)(prow + ks * 32 + 16 + koff);
      }
    }

    // ---- O += P @ V (B fragments from V^T rows) ----
#pragma unroll
    for (int t = 0; t < 4; ++t) {
      const bf16* vtrow = &VTlds[(t * 16 + l15) * 64];
      FragA bv;
      bv.h[0] = *(const v8bf*)(vtrow + koff);
      bv.h[1] = *(const v8bf*)(vtrow + koff + 16);
      o[t] = wmma_bf16(ap[0].v, bv.v, o[t]);
      bv.h[0] = *(const v8bf*)(vtrow + 32 + koff);
      bv.h[1] = *(const v8bf*)(vtrow + 32 + koff + 16);
      o[t] = wmma_bf16(ap[1].v, bv.v, o[t]);
    }
  }

  // ---- finalize: divide by l, store ctx [B,S,D] bf16 ----
#pragma unroll
  for (int i = 0; i < 8; ++i) {
    const float inv = 1.f / lrun[i];
    const int sq = qw + i + hi * 8;
    bf16* dst = ctx + ((size_t)(b_ * SS + sq)) * DD + h_ * DKK;
#pragma unroll
    for (int t = 0; t < 4; ++t)
      dst[t * 16 + l15] = (bf16)(o[t][i] * inv);
  }
}

// ---------------------------------------------------------------------------
extern "C" void kernel_launch(void* const* d_in, const int* in_sizes, int n_in,
                              void* d_out, int out_size, void* d_ws, size_t ws_size,
                              hipStream_t stream) {
  (void)in_sizes; (void)n_in; (void)out_size; (void)ws_size;
  const float* k_in = (const float*)d_in[0];
  const float* q_in = (const float*)d_in[1];
  const float* v_in = (const float*)d_in[2];
  const float* w_q  = (const float*)d_in[3];
  const float* b_q  = (const float*)d_in[4];
  const float* w_k  = (const float*)d_in[5];
  const float* b_k  = (const float*)d_in[6];
  const float* w_v  = (const float*)d_in[7];
  const float* b_v  = (const float*)d_in[8];
  const float* w_o  = (const float*)d_in[9];
  const float* b_o  = (const float*)d_in[10];
  // d_in[11] = mask (causal, handled analytically)

  const size_t ACT = (size_t)BB * SS * DD;  // 4 Mi elements
  const size_t WSZ = (size_t)DD * DD;       // 1 Mi elements
  bf16* qb  = (bf16*)d_ws;
  bf16* kb  = qb  + ACT;
  bf16* vb  = kb  + ACT;
  bf16* wqb = vb  + ACT;
  bf16* wkb = wqb + WSZ;
  bf16* wvb = wkb + WSZ;
  bf16* wob = wvb + WSZ;
  bf16* Qh  = wob + WSZ;
  bf16* Kh  = Qh  + ACT;
  bf16* Vh  = Kh  + ACT;
  bf16* ctx = Vh  + ACT;   // total ~64 MB of workspace

  cvt_f32_to_bf16<<<1024, 256, 0, stream>>>(q_in, qb, (int)ACT);
  cvt_f32_to_bf16<<<1024, 256, 0, stream>>>(k_in, kb, (int)ACT);
  cvt_f32_to_bf16<<<1024, 256, 0, stream>>>(v_in, vb, (int)ACT);
  cvt_f32_to_bf16<<<512, 256, 0, stream>>>(w_q, wqb, (int)WSZ);
  cvt_f32_to_bf16<<<512, 256, 0, stream>>>(w_k, wkb, (int)WSZ);
  cvt_f32_to_bf16<<<512, 256, 0, stream>>>(w_v, wvb, (int)WSZ);
  cvt_f32_to_bf16<<<512, 256, 0, stream>>>(w_o, wob, (int)WSZ);

  dim3 gg(16, 16);  // M/256=16, N/64=16
  proj_gemm<0><<<gg, 256, 0, stream>>>(qb, wqb, b_q, Qh, nullptr);
  proj_gemm<0><<<gg, 256, 0, stream>>>(kb, wkb, b_k, Kh, nullptr);
  proj_gemm<0><<<gg, 256, 0, stream>>>(vb, wvb, b_v, Vh, nullptr);

  attn_kernel<<<dim3(SS / 64, BB * HH), 128, 0, stream>>>(Qh, Kh, Vh, ctx);

  proj_gemm<1><<<gg, 256, 0, stream>>>(ctx, wob, b_o, nullptr, (float*)d_out);
}
